// SBERTaAttention_6090263625871
// MI455X (gfx1250) — matrix operand order repaired and checked by reading, not compiled, not run on hardware
//
#include <hip/hip_runtime.h>
#include <hip/hip_bf16.h>

#define Bsz 4
#define Tlen 2048
#define Dm 768
#define NHh 12
#define DHd 64

typedef _Float16 v16h __attribute__((ext_vector_type(16)));
typedef _Float16 v8h  __attribute__((ext_vector_type(8)));
typedef float    v8f  __attribute__((ext_vector_type(8)));

#if defined(__AMDGCN__) && __has_builtin(__builtin_amdgcn_global_load_async_to_lds_b128)
#define HAVE_ASYNC_LDS 1
typedef int async_v4i __attribute__((vector_size(16)));
typedef __attribute__((address_space(1))) async_v4i* g_v4i_ptr;
typedef __attribute__((address_space(3))) async_v4i* l_v4i_ptr;
#else
#define HAVE_ASYNC_LDS 0
#endif

__device__ __forceinline__ v16h cat8(v8h lo, v8h hi) {
  return __builtin_shufflevector(lo, hi, 0,1,2,3,4,5,6,7,8,9,10,11,12,13,14,15);
}

__device__ __forceinline__ v8f wmma16(v16h a, v16h b, v8f c) {
  return __builtin_amdgcn_wmma_f32_16x16x32_f16(false, a, false, b, (short)0, c, false, false);
}

// A-fragment (16x32 f16, M x K): lane l -> row = l&15, h = l>>4,
// element e -> k = h*16 + e. Two contiguous 16B loads.
__device__ __forceinline__ v16h load_afrag(const _Float16* rowptr_k0, int h) {
  const _Float16* p0 = rowptr_k0 + h * 16;
  v8h lo = *(const v8h*)(p0);
  v8h hi = *(const v8h*)(p0 + 8);
  return cat8(lo, hi);
}

__device__ __forceinline__ void wait_async0() {
#if HAVE_ASYNC_LDS
#if __has_builtin(__builtin_amdgcn_s_wait_asynccnt)
  __builtin_amdgcn_s_wait_asynccnt(0);
#else
  asm volatile("s_wait_asynccnt 0" ::: "memory");
#endif
#endif
}

// ---------------- elementwise converts ----------------
__global__ void cvt_f32_f16(const float* __restrict__ src, _Float16* __restrict__ dst, int n) {
  int i = blockIdx.x * blockDim.x + threadIdx.x;
  if (i < n) dst[i] = (_Float16)src[i];
}

// pc[b,h,t,l] = sum_k p[b,t,k] * compat[h,k,l]   (K = 2)
__global__ void pc_kernel(const float* __restrict__ p, const float* __restrict__ compat,
                          float* __restrict__ pc) {
  int i = blockIdx.x * blockDim.x + threadIdx.x;
  const int total = Bsz * NHh * Tlen * 2;
  if (i >= total) return;
  int l = i & 1;
  int t = (i >> 1) % Tlen;
  int h = (i >> 1) / Tlen % NHh;
  int b = i / (2 * Tlen * NHh);
  float p0 = p[((size_t)b * Tlen + t) * 2 + 0];
  float p1 = p[((size_t)b * Tlen + t) * 2 + 1];
  pc[i] = p0 * compat[(h * 2 + 0) * 2 + l] + p1 * compat[(h * 2 + 1) * 2 + l];
}

// ---------------- projection GEMM, wave = 32(M) x 64(N) tile ----------------
// Out[b,head,t,dh] = scale * (H @ W^T); W is (out,in) row-major as in nn.Linear.
__global__ void proj_gemm(const _Float16* __restrict__ A, const _Float16* __restrict__ W,
                          _Float16* __restrict__ Out, float scale) {
  const int lane = threadIdx.x & 31;
  const int wave = threadIdx.x >> 5;
  const int tile = blockIdx.x * (blockDim.x >> 5) + wave;
  const int NB = Dm / 64;                 // 12 n-blocks (= head index)
  const int mb = tile / NB, nb = tile % NB;
  const int r = lane & 15, h = lane >> 4;

  const _Float16* arow0 = A + (size_t)(mb * 32 + r) * Dm;       // M rows 0..15
  const _Float16* arow1 = arow0 + (size_t)16 * Dm;              // M rows 16..31
  const _Float16* wrow0 = W + (size_t)(nb * 64 + r) * Dm;       // B-frag col n = r
  v8f acc[2][4] = {};
  for (int k0 = 0; k0 < Dm; k0 += 32) {
    __builtin_prefetch(arow0 + k0 + 256, 0, 1);
    __builtin_prefetch(arow1 + k0 + 256, 0, 1);
    v16h a0 = load_afrag(arow0 + k0, h);
    v16h a1 = load_afrag(arow1 + k0, h);
    #pragma unroll
    for (int nt = 0; nt < 4; ++nt) {
      v16h b = *(const v16h*)(wrow0 + (size_t)nt * 16 * Dm + k0 + h * 16);
      acc[0][nt] = wmma16(a0, b, acc[0][nt]);
      acc[1][nt] = wmma16(a1, b, acc[1][nt]);
    }
  }
  // C layout: lane holds (m = v + 8h, n = r); scatter into [B, NH, T, DH]
  #pragma unroll
  for (int mi = 0; mi < 2; ++mi)
    #pragma unroll
    for (int nt = 0; nt < 4; ++nt)
      #pragma unroll
      for (int v = 0; v < 8; ++v) {
        int row = mb * 32 + mi * 16 + v + 8 * h;
        int col = nb * 64 + nt * 16 + r;
        int bb = row >> 11;               // / Tlen
        int t  = row & (Tlen - 1);
        int head = col >> 6;              // / DHd
        int dh   = col & (DHd - 1);
        Out[(((size_t)bb * NHh + head) * Tlen + t) * DHd + dh] =
            (_Float16)(acc[mi][nt][v] * scale);
      }
}

// ---------------- flash attention ----------------
#define ATT_WAVES 4
#define WAVE_LDS  2560   // halves: 16x32 P buffer (512) + 32x64 V tile (2048)

__global__ void attn_kernel(const _Float16* __restrict__ Qh, const _Float16* __restrict__ Kh,
                            const _Float16* __restrict__ Vh, const float* __restrict__ p,
                            const float* __restrict__ s, const float* __restrict__ pc,
                            const float* __restrict__ gamma, _Float16* __restrict__ Ctx) {
  __shared__ __attribute__((aligned(64))) _Float16 smem[ATT_WAVES * WAVE_LDS];
  const int lane = threadIdx.x & 31;
  const int wave = threadIdx.x >> 5;
  const int wid  = blockIdx.x * ATT_WAVES + wave;
  const int qtile = wid % (Tlen / 16);
  const int bh    = wid / (Tlen / 16);
  const int b = bh / NHh, head = bh % NHh;
  const int r = lane & 15, h = lane >> 4;

  _Float16* Pb = smem + wave * WAVE_LDS;  // P tile, row-major 16x32
  _Float16* Vb = Pb + 512;                // V tile (layout depends on staging path)

  const size_t headbase = ((size_t)b * NHh + head) * Tlen;

  // Q A-fragments (softmax scale folded in at projection)
  const _Float16* qrow = Qh + (headbase + qtile * 16 + r) * DHd;
  v16h aq0 = load_afrag(qrow + 0,  h);
  v16h aq1 = load_afrag(qrow + 32, h);

  float pc0[8], pc1[8];
  #pragma unroll
  for (int v = 0; v < 8; ++v) {
    size_t i = headbase + qtile * 16 + v + 8 * h;
    pc0[v] = pc[i * 2 + 0];
    pc1[v] = pc[i * 2 + 1];
  }
  const float g = gamma[0];

  float mrow[8], lrow[8], corrv[8];
  v8f oacc[4] = {};
  #pragma unroll
  for (int v = 0; v < 8; ++v) { mrow[v] = -1e30f; lrow[v] = 0.0f; }

  for (int jj = 0; jj < Tlen; jj += 32) {
    // prefetch next K/V tiles
    if (jj + 32 < Tlen) {
      __builtin_prefetch(Kh + (headbase + jj + 32 + lane) * DHd, 0, 1);
      __builtin_prefetch(Vh + (headbase + jj + 32 + lane) * DHd, 0, 1);
    }

    // ---- stage 32x64 V tile into LDS ----
    const _Float16* vrow = Vh + (headbase + jj + lane) * DHd;
#if HAVE_ASYNC_LDS
    // async global->LDS, row-major Vb[key*64 + dh]; 16B per lane per issue
    #pragma unroll
    for (int u = 0; u < 8; ++u) {
      __builtin_amdgcn_global_load_async_to_lds_b128(
          (g_v4i_ptr)(vrow + u * 8),
          (l_v4i_ptr)(Vb + lane * DHd + u * 8),
          0, 0);
    }
#else
    // fallback: transpose on store, Vb[dh*32 + key]
    #pragma unroll
    for (int c0 = 0; c0 < DHd; c0 += 8) {
      v8h vv = *(const v8h*)(vrow + c0);
      #pragma unroll
      for (int u = 0; u < 8; ++u) Vb[(c0 + u) * 32 + lane] = vv[u];
    }
#endif

    float sv[2][8];
    #pragma unroll
    for (int half = 0; half < 2; ++half) {
      const int j0 = jj + half * 16;
      // K B-fragments: n = r -> key row j0+r; k(e) = f*32 + h*16 + e (contiguous)
      const _Float16* krow = Kh + (headbase + j0 + r) * DHd;
      v16h bk0 = *(const v16h*)(krow + 0  + h * 16);
      v16h bk1 = *(const v16h*)(krow + 32 + h * 16);
      v8f sacc = {};
      sacc = wmma16(aq0, bk0, sacc);
      sacc = wmma16(aq1, bk1, sacc);
      const float pj0 = p[((size_t)b * Tlen + j0 + r) * 2 + 0];
      const float pj1 = p[((size_t)b * Tlen + j0 + r) * 2 + 1];
      const float sj  = s[(size_t)b * Tlen + j0 + r];
      #pragma unroll
      for (int v = 0; v < 8; ++v)
        sv[half][v] = sacc[v] + pc0[v] * pj0 + pc1[v] * pj1 + g * sj;
    }

    // online softmax (row reductions across the 16-lane column groups)
    #pragma unroll
    for (int v = 0; v < 8; ++v) {
      float tm = fmaxf(sv[0][v], sv[1][v]);
      #pragma unroll
      for (int mask = 1; mask < 16; mask <<= 1)
        tm = fmaxf(tm, __shfl_xor(tm, mask, 32));
      float mn = fmaxf(mrow[v], tm);
      float corr = __expf(mrow[v] - mn);
      float p0 = __expf(sv[0][v] - mn);
      float p1 = __expf(sv[1][v] - mn);
      float ts = p0 + p1;
      #pragma unroll
      for (int mask = 1; mask < 16; mask <<= 1)
        ts += __shfl_xor(ts, mask, 32);
      lrow[v] = lrow[v] * corr + ts;
      mrow[v] = mn;
      corrv[v] = corr;
      sv[0][v] = p0;
      sv[1][v] = p1;
    }
    #pragma unroll
    for (int d = 0; d < 4; ++d)
      #pragma unroll
      for (int v = 0; v < 8; ++v) oacc[d][v] *= corrv[v];

    // P (C-layout) -> LDS row-major 16x32 -> A-fragment (wave-private, LDS in-order)
    #pragma unroll
    for (int half = 0; half < 2; ++half)
      #pragma unroll
      for (int v = 0; v < 8; ++v)
        Pb[(v + 8 * h) * 32 + half * 16 + r] = (_Float16)sv[half][v];
    v16h ap = load_afrag(Pb + r * 32, h);

    // make async V staging visible before LDS reads of Vb
    wait_async0();

    // PV: 4 dh-chunks
    #pragma unroll
    for (int d = 0; d < 4; ++d) {
      v16h bv;
#if HAVE_ASYNC_LDS
      // row-major Vb: element e -> Vb[(h*16+e)*64 + d*16 + r] (strided gather)
      #pragma unroll
      for (int e = 0; e < 16; ++e) bv[e] = Vb[(h * 16 + e) * DHd + d * 16 + r];
#else
      // transposed Vb: element e -> Vb[(d*16+r)*32 + h*16 + e] (contiguous)
      bv = *(const v16h*)(Vb + (d * 16 + r) * 32 + h * 16);
#endif
      oacc[d] = wmma16(ap, bv, oacc[d]);
    }
  }

  // normalize + write merged context [B*T, D] as f16
  #pragma unroll
  for (int d = 0; d < 4; ++d)
    #pragma unroll
    for (int v = 0; v < 8; ++v) {
      int m = v + 8 * h;
      size_t row = (size_t)b * Tlen + qtile * 16 + m;
      int col = head * DHd + d * 16 + r;
      Ctx[row * Dm + col] = (_Float16)(oacc[d][v] / lrow[v]);
    }
}

// ---------------- output projection: out = Ctx @ Wo^T + bo (f32), 32x64 per wave ----------------
__global__ void out_gemm(const _Float16* __restrict__ A, const _Float16* __restrict__ W,
                         const float* __restrict__ bias, float* __restrict__ Out) {
  const int lane = threadIdx.x & 31;
  const int wave = threadIdx.x >> 5;
  const int tile = blockIdx.x * (blockDim.x >> 5) + wave;
  const int NB = Dm / 64;
  const int mb = tile / NB, nb = tile % NB;
  const int r = lane & 15, h = lane >> 4;

  const _Float16* arow0 = A + (size_t)(mb * 32 + r) * Dm;
  const _Float16* arow1 = arow0 + (size_t)16 * Dm;
  const _Float16* wrow0 = W + (size_t)(nb * 64 + r) * Dm;
  v8f acc[2][4] = {};
  for (int k0 = 0; k0 < Dm; k0 += 32) {
    __builtin_prefetch(arow0 + k0 + 256, 0, 1);
    __builtin_prefetch(arow1 + k0 + 256, 0, 1);
    v16h a0 = load_afrag(arow0 + k0, h);
    v16h a1 = load_afrag(arow1 + k0, h);
    #pragma unroll
    for (int nt = 0; nt < 4; ++nt) {
      v16h b = *(const v16h*)(wrow0 + (size_t)nt * 16 * Dm + k0 + h * 16);
      acc[0][nt] = wmma16(a0, b, acc[0][nt]);
      acc[1][nt] = wmma16(a1, b, acc[1][nt]);
    }
  }
  #pragma unroll
  for (int nt = 0; nt < 4; ++nt) {
    const int col = nb * 64 + nt * 16 + r;
    const float bc = bias[col];
    #pragma unroll
    for (int mi = 0; mi < 2; ++mi)
      #pragma unroll
      for (int v = 0; v < 8; ++v) {
        int row = mb * 32 + mi * 16 + v + 8 * h;
        Out[(size_t)row * Dm + col] = acc[mi][nt][v] + bc;
      }
  }
}

extern "C" void kernel_launch(void* const* d_in, const int* in_sizes, int n_in,
                              void* d_out, int out_size, void* d_ws, size_t ws_size,
                              hipStream_t stream) {
  const float* H      = (const float*)d_in[0];
  const float* p      = (const float*)d_in[1];
  const float* s      = (const float*)d_in[2];
  const float* wq     = (const float*)d_in[3];
  const float* wk     = (const float*)d_in[4];
  const float* wv     = (const float*)d_in[5];
  const float* wo     = (const float*)d_in[6];
  const float* bo     = (const float*)d_in[7];
  const float* compat = (const float*)d_in[8];
  const float* gamma  = (const float*)d_in[9];
  float* out = (float*)d_out;

  const size_t nH  = (size_t)Bsz * Tlen * Dm;   // 6,291,456
  const size_t nW  = (size_t)Dm * Dm;           // 589,824
  const size_t nPC = (size_t)Bsz * NHh * Tlen * 2;

  char* ws = (char*)d_ws;
  size_t off = 0;
  auto take = [&](size_t bytes) -> void* {
    void* q = ws + off;
    off = (off + bytes + 255) & ~(size_t)255;
    return q;
  };
  _Float16* Hh  = (_Float16*)take(nH * 2);
  _Float16* Wqh = (_Float16*)take(nW * 2);
  _Float16* Wkh = (_Float16*)take(nW * 2);
  _Float16* Wvh = (_Float16*)take(nW * 2);
  _Float16* Woh = (_Float16*)take(nW * 2);
  _Float16* Qh  = (_Float16*)take(nH * 2);
  _Float16* Kh  = (_Float16*)take(nH * 2);
  _Float16* Vh  = (_Float16*)take(nH * 2);
  _Float16* Ctx = (_Float16*)take(nH * 2);
  float*    pcb = (float*)take(nPC * 4);

  cvt_f32_f16<<<(int)((nH + 255) / 256), 256, 0, stream>>>(H,  Hh,  (int)nH);
  cvt_f32_f16<<<(int)((nW + 255) / 256), 256, 0, stream>>>(wq, Wqh, (int)nW);
  cvt_f32_f16<<<(int)((nW + 255) / 256), 256, 0, stream>>>(wk, Wkh, (int)nW);
  cvt_f32_f16<<<(int)((nW + 255) / 256), 256, 0, stream>>>(wv, Wvh, (int)nW);
  cvt_f32_f16<<<(int)((nW + 255) / 256), 256, 0, stream>>>(wo, Woh, (int)nW);

  // GEMM: M = 8192 -> 256 m-blocks(32), N = 768 -> 12 n-blocks(64); 4 waves/block
  const int gemm_blocks = (256 * 12) / 4;       // 768, exact
  proj_gemm<<<gemm_blocks, 128, 0, stream>>>(Hh, Wqh, Qh, 0.125f); // fold 1/sqrt(64)
  proj_gemm<<<gemm_blocks, 128, 0, stream>>>(Hh, Wkh, Kh, 1.0f);
  proj_gemm<<<gemm_blocks, 128, 0, stream>>>(Hh, Wvh, Vh, 1.0f);

  pc_kernel<<<(int)((nPC + 255) / 256), 256, 0, stream>>>(p, compat, pcb);

  // attention: B*NH*(T/16) = 6144 waves, 4 waves/block -> 1536 blocks, exact
  attn_kernel<<<1536, 128, 0, stream>>>(Qh, Kh, Vh, p, s, pcb, gamma, Ctx);

  out_gemm<<<gemm_blocks, 128, 0, stream>>>(Ctx, Woh, bo, out);
}
